// Gen_criterion_59150289600617
// MI455X (gfx1250) — matrix operand order repaired and checked by reading, not compiled, not run on hardware
//
#include <hip/hip_runtime.h>

#define N_ROWS 1024
#define DIM    8192

typedef _Float16 v8h  __attribute__((ext_vector_type(8)));
typedef _Float16 v16h __attribute__((ext_vector_type(16)));
typedef float    v8f  __attribute__((ext_vector_type(8)));

// WGP-scope prefetch (scope field 0 = default): pulls line into ALL cache
// levels on miss, unlike __builtin_prefetch which lowers to SE/SYS scope
// (GL2-only, useless when the operand set already resides in L2).
__device__ __forceinline__ void prefetch_wgp(const void* p) {
    asm volatile("global_prefetch_b8 %0, off" :: "v"(p));
}

// acc[0] = mse_sum, acc[1] = diag_sum, acc[2] = total_sum
__global__ void zero_accum_kernel(float* acc) {
    if (threadIdx.x < 8) acc[threadIdx.x] = 0.0f;
}

// ---- loss1: mean((pred-eeg)^2), bandwidth bound, float4 + wave reduce ----
__global__ __launch_bounds__(256) void mse_kernel(const float* __restrict__ pred,
                                                  const float* __restrict__ eeg,
                                                  float* acc) {
    const size_t total = (size_t)N_ROWS * DIM / 4;
    size_t i      = (size_t)blockIdx.x * blockDim.x + threadIdx.x;
    size_t stride = (size_t)gridDim.x * blockDim.x;
    const float4* p4 = (const float4*)pred;
    const float4* e4 = (const float4*)eeg;
    float s = 0.0f;
    for (; i < total; i += stride) {
        float4 a = p4[i], b = e4[i];
        float dx = a.x - b.x, dy = a.y - b.y, dz = a.z - b.z, dw = a.w - b.w;
        s += dx * dx + dy * dy + dz * dz + dw * dw;
    }
    for (int off = 16; off; off >>= 1) s += __shfl_xor(s, off, 32);
    if ((threadIdx.x & 31) == 0) atomicAdd(&acc[0], s);
}

// ---- rank transform: bitonic sort row in LDS, binary-search each element's
// rank, emit normalized rank as f16. Ranks are a permutation of 0..D-1 so
// mean and norm are compile-time constants. One block (1024 thr) per row. ----
__global__ __launch_bounds__(1024) void rank_kernel(const float* __restrict__ pred,
                                                    const float* __restrict__ eeg,
                                                    _Float16* __restrict__ rp,
                                                    _Float16* __restrict__ re_out) {
    __shared__ float s[DIM];   // 32 KB
    const int row = blockIdx.x;
    const float*    src = (row < N_ROWS) ? (pred + (size_t)row * DIM)
                                         : (eeg + (size_t)(row - N_ROWS) * DIM);
    _Float16*       dst = (row < N_ROWS) ? (rp + (size_t)row * DIM)
                                         : (re_out + (size_t)(row - N_ROWS) * DIM);
    const int tid = threadIdx.x;

    float xv[DIM / 1024];
    #pragma unroll
    for (int e = 0; e < DIM / 1024; ++e) {
        float v = src[e * 1024 + tid];
        xv[e] = v;
        s[e * 1024 + tid] = v;
    }
    __syncthreads();

    // bitonic sort, ascending
    for (unsigned k = 2; k <= DIM; k <<= 1) {
        for (unsigned j = k >> 1; j > 0; j >>= 1) {
            #pragma unroll
            for (int e = 0; e < DIM / 1024; ++e) {
                unsigned i   = (unsigned)(e * 1024 + tid);
                unsigned ixj = i ^ j;
                if (ixj > i) {
                    float a = s[i], b = s[ixj];
                    bool up = ((i & k) == 0);
                    if ((a > b) == up) { s[i] = b; s[ixj] = a; }
                }
            }
            __syncthreads();
        }
    }

    // rank = #elements < x (no ties); constants: mean=(D-1)/2, 1/||r-mean||
    const float cmean   = 4095.5f;
    const float invnorm = 4.6720235e-06f;   // 1/sqrt(D*(D*D-1)/12), D=8192
    #pragma unroll
    for (int e = 0; e < DIM / 1024; ++e) {
        float x = xv[e];
        int lo = 0, hi = DIM;
        while (lo < hi) { int mid = (lo + hi) >> 1; if (s[mid] < x) lo = mid + 1; else hi = mid; }
        dst[e * 1024 + tid] = (_Float16)(((float)lo - cmean) * invnorm);
    }
}

// ---- corr = rp @ re^T via v_wmma_f32_16x16x32_f16, fused diag/offdiag
// reduction. One wave per 16x16 output tile, 8 waves / block, 512 blocks.
// K unrolled x4 with dual accumulators to break WMMA->WMMA RAW chains. ----
__global__ __launch_bounds__(256) void corr_kernel(const _Float16* __restrict__ rp,
                                                   const _Float16* __restrict__ re_in,
                                                   float* acc) {
    const int wave = threadIdx.x >> 5;
    const int lane = threadIdx.x & 31;
    const int tile = blockIdx.x * 8 + wave;
    const int tm = tile >> 6;          // 64x64 tiles over 1024x1024
    const int tn = tile & 63;
    const int hl   = lane >> 4;        // half-wave selector
    const int mloc = lane & 15;

    // A: rp row (tm*16+mloc), K chunks {hl*8 .. hl*8+7} and {16+hl*8 ..}
    const _Float16* pa = rp    + (size_t)(tm * 16 + mloc) * DIM + hl * 8;
    // B = re^T: column n = re row (tn*16+mloc), K chunks {hl*8 .. hl*8+7} and {16+hl*8 ..}
    const _Float16* pb = re_in + (size_t)(tn * 16 + mloc) * DIM + hl * 8;

    v8f c0 = {}, c1 = {};
    for (int k0 = 0; k0 < DIM; k0 += 128) {
        // WGP-scope prefetch of the next 256B window of both operand rows
        prefetch_wgp(pa + k0 + 128);
        prefetch_wgp(pb + k0 + 128);
        #pragma unroll
        for (int u = 0; u < 4; ++u) {
            const int k = k0 + u * 32;
            v8h alo = *(const v8h*)(pa + k);
            v8h ahi = *(const v8h*)(pa + k + 16);
            v16h a  = __builtin_shufflevector(alo, ahi,
                        0,1,2,3,4,5,6,7,8,9,10,11,12,13,14,15);
            v8h blo = *(const v8h*)(pb + k);
            v8h bhi = *(const v8h*)(pb + k + 16);
            v16h b  = __builtin_shufflevector(blo, bhi,
                        0,1,2,3,4,5,6,7,8,9,10,11,12,13,14,15);
            if (u & 1)
                c1 = __builtin_amdgcn_wmma_f32_16x16x32_f16(
                        false, a, false, b, (short)0, c1, false, false);
            else
                c0 = __builtin_amdgcn_wmma_f32_16x16x32_f16(
                        false, a, false, b, (short)0, c0, false, false);
        }
    }
    v8f c = c0 + c1;

    // tile total
    float tsum = 0.0f;
    #pragma unroll
    for (int j = 0; j < 8; ++j) tsum += c[j];
    // diagonal contribution: VGPR j holds C[j + hl*8][mloc]
    float dsum = 0.0f;
    if (tm == tn) {
        int j = mloc - hl * 8;
        if (j >= 0 && j < 8) dsum = c[j];
    }
    for (int off = 16; off; off >>= 1) {
        tsum += __shfl_xor(tsum, off, 32);
        dsum += __shfl_xor(dsum, off, 32);
    }
    if (lane == 0) {
        atomicAdd(&acc[2], tsum);
        if (tm == tn) atomicAdd(&acc[1], dsum);
    }
}

__global__ void finalize_kernel(const float* acc, float* out) {
    if (threadIdx.x == 0 && blockIdx.x == 0) {
        float mse = acc[0] / ((float)N_ROWS * (float)DIM);
        float pos = acc[1] / (float)N_ROWS;
        float neg = (acc[2] - acc[1]) / ((float)N_ROWS * (float)(N_ROWS - 1));
        out[0] = mse + 1.0f - pos + neg;
    }
}

extern "C" void kernel_launch(void* const* d_in, const int* in_sizes, int n_in,
                              void* d_out, int out_size, void* d_ws, size_t ws_size,
                              hipStream_t stream) {
    const float* pred = (const float*)d_in[0];
    const float* eeg  = (const float*)d_in[1];
    float* out = (float*)d_out;

    char* ws = (char*)d_ws;
    float*    acc = (float*)ws;                         // 3 accumulators
    _Float16* rp  = (_Float16*)(ws + 256);              // 16 MB
    _Float16* re_ = rp + (size_t)N_ROWS * DIM;          // 16 MB

    zero_accum_kernel<<<1, 32, 0, stream>>>(acc);
    mse_kernel<<<1024, 256, 0, stream>>>(pred, eeg, acc);
    rank_kernel<<<2 * N_ROWS, 1024, 0, stream>>>(pred, eeg, rp, re_);
    corr_kernel<<<512, 256, 0, stream>>>(rp, re_, acc);
    finalize_kernel<<<1, 32, 0, stream>>>(acc, out);
}
